// Head_1116691497433
// MI455X (gfx1250) — matrix operand order repaired and checked by reading, not compiled, not run on hardware
//
#include <hip/hip_runtime.h>
#include <hip/hip_bf16.h>

// ---------------------------------------------------------------------------
// Problem constants (fixed by the reference)
// ---------------------------------------------------------------------------
#define NN 262144
#define GG 512
#define HH 16
#define EPSF 1e-5f

typedef __attribute__((ext_vector_type(16))) _Float16 v16h;
typedef __attribute__((ext_vector_type(8)))  float    v8f;

// Workspace layout (floats)
#define WS_SCALE   0         // G*64  : per-(g,c) scale  (gamma*rstd)      [pass1: sum x]
#define WS_BIAS    32768     // G*64  : per-(g,c) bias   (beta-mean*scale) [pass1: sum x^2]
#define WS_CNT     65536     // G     : node count per graph
#define WS_GMAX    66048     // G*3   : per-graph logit max
#define WS_GSUM    67584     // G*3   : per-graph sum(exp)
#define WS_ACC     69120     // G*21  : out10(3) out01(3) out11(9) m1m2(6)
#define WS_LOGITS  79872     // N*3   : logits, then exp(logit-max)
#define WS_TOTAL   866304    // floats (~3.47 MB)

__device__ __forceinline__ void atomicMaxFloat(float* addr, float v) {
    // Sign-split trick: int max for >=0, uint min for <0 (init must be -FLT_MAX).
    if (v >= 0.0f) atomicMax((int*)addr, __float_as_int(v));
    else           atomicMin((unsigned int*)addr, __float_as_uint(v));
}

// ---------------------------------------------------------------------------
// K0: zero workspace, init gmax to -FLT_MAX
// ---------------------------------------------------------------------------
__global__ void k0_init(float* ws) {
    int t = blockIdx.x * blockDim.x + threadIdx.x;
    if (t >= WS_TOTAL) return;
    ws[t] = (t >= WS_GMAX && t < WS_GMAX + GG * 3) ? -3.402823466e38f : 0.0f;
}

// ---------------------------------------------------------------------------
// K1: per-graph, per-channel sum / sumsq of x00, plus counts.
// thread = (node, channel-quad); float4 loads are fully coalesced.
// ---------------------------------------------------------------------------
__global__ void k1_stats(const float* __restrict__ x00,
                         const int* __restrict__ gid, float* ws) {
    int t = blockIdx.x * blockDim.x + threadIdx.x;
    if (t >= NN * 16) return;
    int n = t >> 4, q = t & 15;
    float4 v = ((const float4*)x00)[n * 16 + q];
    int g = gid[n];
    float* s  = ws + WS_SCALE + g * 64 + q * 4;
    float* s2 = ws + WS_BIAS  + g * 64 + q * 4;
    atomicAdd(&s[0], v.x);  atomicAdd(&s2[0], v.x * v.x);
    atomicAdd(&s[1], v.y);  atomicAdd(&s2[1], v.y * v.y);
    atomicAdd(&s[2], v.z);  atomicAdd(&s2[2], v.z * v.z);
    atomicAdd(&s[3], v.w);  atomicAdd(&s2[3], v.w * v.w);
    if (q == 0) atomicAdd(ws + WS_CNT + g, 1.0f);
}

// ---------------------------------------------------------------------------
// K2: fold gamma/beta into per-(g,c) affine:  xn = x*scale + bias
// ---------------------------------------------------------------------------
__global__ void k2_finalize(const float* __restrict__ gamma,
                            const float* __restrict__ beta, float* ws) {
    int t = blockIdx.x * blockDim.x + threadIdx.x;
    if (t >= GG * 64) return;
    int g = t >> 6, c = t & 63;
    float cc   = fmaxf(ws[WS_CNT + g], 1.0f);
    float mean = ws[WS_SCALE + t] / cc;
    float var  = ws[WS_BIAS + t] / cc - mean * mean;
    float sc   = rsqrtf(var + EPSF) * gamma[c];
    ws[WS_SCALE + t] = sc;
    ws[WS_BIAS + t]  = beta[c] - mean * sc;
}

// ---------------------------------------------------------------------------
// K3: logits = xn @ lin_w^T + lin_b via v_wmma_f32_16x16x32_f16.
// One wave per 16-node tile; A = xn tile (16x64 f16, two K=32 steps),
// B = lin_w^T (64x16, cols 3..15 zero).
//
// A layout (ISA 16-bit A 16x32): lane L<16 -> M=L, K {0..7,16..23};
// lane L+16 -> M=L, K {8..15,24..31}. Each 8-element K-group is contiguous
// in memory -> float4 (b128) loads of x00/scale/bias.
// B layout (matches A): lanes 0-15 hold K {0..7,16..23} for N=lane;
// lanes 16-31 hold K {8..15,24..31} for N=lane-16.
// B rows come from lin_w staged in LDS; row index clamped (no divergence),
// zeroed by a 0/1 mask for N>=3.
// C layout: lanes 0-15 row r=vgpr, col=lane; lanes 16-31 row 8+r, col=lane-16.
// ---------------------------------------------------------------------------
__global__ void k3_logits_wmma(const float* __restrict__ x00,
                               const int* __restrict__ gid,
                               const float* __restrict__ lin_w,
                               const float* __restrict__ lin_b,
                               float* ws) {
    __shared__ float lw[3 * 64];
    for (int i = threadIdx.x; i < 3 * 64; i += blockDim.x) lw[i] = lin_w[i];
    __syncthreads();

    const int lane = threadIdx.x & 31;
    const int tile = blockIdx.x * (blockDim.x >> 5) + (threadIdx.x >> 5);
    const int nb   = tile * 16;
    const int m    = lane & 15;            // node-in-tile / B column this lane owns
    const int mr   = (m < 3) ? m : 0;      // clamped B row (keeps loads in-bounds)
    const float mmask = (m < 3) ? 1.0f : 0.0f;
    const int node = nb + m;
    const int g    = gid[node];
    const int kh   = lane >> 4;

    const float4* xp  = (const float4*)(x00 + node * 64);
    const float4* scp = (const float4*)(ws + WS_SCALE + g * 64);
    const float4* bip = (const float4*)(ws + WS_BIAS  + g * 64);

    v8f c = {};
    #pragma unroll
    for (int kb = 0; kb < 64; kb += 32) {
        const int q = (kb + (kh << 3)) >> 2;   // quad index of first K-group
        float4 xa0 = xp[q],     xa1 = xp[q + 1];
        float4 xa2 = xp[q + 4], xa3 = xp[q + 5];
        float4 s0  = scp[q],     s1 = scp[q + 1];
        float4 s2  = scp[q + 4], s3 = scp[q + 5];
        float4 b0  = bip[q],     b1 = bip[q + 1];
        float4 b2  = bip[q + 4], b3 = bip[q + 5];

        v16h a;
        a[0]  = (_Float16)(xa0.x * s0.x + b0.x);
        a[1]  = (_Float16)(xa0.y * s0.y + b0.y);
        a[2]  = (_Float16)(xa0.z * s0.z + b0.z);
        a[3]  = (_Float16)(xa0.w * s0.w + b0.w);
        a[4]  = (_Float16)(xa1.x * s1.x + b1.x);
        a[5]  = (_Float16)(xa1.y * s1.y + b1.y);
        a[6]  = (_Float16)(xa1.z * s1.z + b1.z);
        a[7]  = (_Float16)(xa1.w * s1.w + b1.w);
        a[8]  = (_Float16)(xa2.x * s2.x + b2.x);
        a[9]  = (_Float16)(xa2.y * s2.y + b2.y);
        a[10] = (_Float16)(xa2.z * s2.z + b2.z);
        a[11] = (_Float16)(xa2.w * s2.w + b2.w);
        a[12] = (_Float16)(xa3.x * s3.x + b3.x);
        a[13] = (_Float16)(xa3.y * s3.y + b3.y);
        a[14] = (_Float16)(xa3.z * s3.z + b3.z);
        a[15] = (_Float16)(xa3.w * s3.w + b3.w);

        // B fragment from LDS (ds_load_b128 x4), masked for N>=3
        const int bbase = kb + (kh << 3);
        const float4* lwp = (const float4*)(lw + mr * 64 + bbase);
        float4 w0 = lwp[0], w1 = lwp[1], w2 = lwp[4], w3 = lwp[5];
        v16h b;
        b[0]  = (_Float16)(w0.x * mmask);
        b[1]  = (_Float16)(w0.y * mmask);
        b[2]  = (_Float16)(w0.z * mmask);
        b[3]  = (_Float16)(w0.w * mmask);
        b[4]  = (_Float16)(w1.x * mmask);
        b[5]  = (_Float16)(w1.y * mmask);
        b[6]  = (_Float16)(w1.z * mmask);
        b[7]  = (_Float16)(w1.w * mmask);
        b[8]  = (_Float16)(w2.x * mmask);
        b[9]  = (_Float16)(w2.y * mmask);
        b[10] = (_Float16)(w2.z * mmask);
        b[11] = (_Float16)(w2.w * mmask);
        b[12] = (_Float16)(w3.x * mmask);
        b[13] = (_Float16)(w3.y * mmask);
        b[14] = (_Float16)(w3.z * mmask);
        b[15] = (_Float16)(w3.w * mmask);

        c = __builtin_amdgcn_wmma_f32_16x16x32_f16(false, a, false, b,
                                                   (short)0, c, false, false);
    }

    const int col = lane & 15;
    const int rowBase = (lane >> 4) * 8;
    if (col < 3) {
        float lb = lin_b[col];
        float* logits = ws + WS_LOGITS;
        float* gmax   = ws + WS_GMAX;
        #pragma unroll
        for (int r = 0; r < 8; ++r) {
            int n2 = nb + rowBase + r;
            float lv = c[r] + lb;
            logits[n2 * 3 + col] = lv;
            atomicMaxFloat(&gmax[gid[n2] * 3 + col], lv);
        }
    }
}

// ---------------------------------------------------------------------------
// K4: e = exp(logit - gmax); accumulate per-graph denominator
// ---------------------------------------------------------------------------
__global__ void k4_exp(const int* __restrict__ gid, float* ws) {
    int t = blockIdx.x * blockDim.x + threadIdx.x;
    if (t >= NN * 3) return;
    int n = t / 3, cidx = t - n * 3;
    int g = gid[n];
    float e = __expf(ws[WS_LOGITS + t] - ws[WS_GMAX + g * 3 + cidx]);
    ws[WS_LOGITS + t] = e;
    atomicAdd(ws + WS_GSUM + g * 3 + cidx, e);
}

// ---------------------------------------------------------------------------
// K5: per node: softmax weights, pre-contract features over H with
// w10/w01/w11, then 21 atomic adds into per-graph accumulators.
// ---------------------------------------------------------------------------
__global__ void k5_accumulate(const float* __restrict__ x10,
                              const float* __restrict__ x01,
                              const float* __restrict__ x11,
                              const float* __restrict__ pos,
                              const int* __restrict__ gid,
                              const float* __restrict__ w10,
                              const float* __restrict__ w01,
                              const float* __restrict__ w11,
                              float* ws) {
    int n = blockIdx.x * blockDim.x + threadIdx.x;
    if (n >= NN) return;
    int g = gid[n];
    const float* e  = ws + WS_LOGITS + n * 3;
    const float* gs = ws + WS_GSUM + g * 3;
    float w0 = e[0] / gs[0], w1 = e[1] / gs[1], w2 = e[2] / gs[2];

    const float* xa = x10 + n * 48;
    const float* xb = x01 + n * 48;
    const float* xc = x11 + n * 144;
    __builtin_prefetch(xc, 0, 1);            // global_prefetch_b8
    __builtin_prefetch(xc + 64, 0, 1);

    float p10[3] = {0.f, 0.f, 0.f}, p01[3] = {0.f, 0.f, 0.f};
    float p11[9] = {0.f, 0.f, 0.f, 0.f, 0.f, 0.f, 0.f, 0.f, 0.f};
    #pragma unroll
    for (int h = 0; h < HH; ++h) {
        float wa = w10[h], wb = w01[h], wc = w11[h];
        #pragma unroll
        for (int d = 0; d < 3; ++d) {
            p10[d] += wa * xa[h * 3 + d];
            p01[d] += wb * xb[h * 3 + d];
        }
        #pragma unroll
        for (int d = 0; d < 9; ++d) p11[d] += wc * xc[h * 9 + d];
    }
    float* ag = ws + WS_ACC + g * 21;
    #pragma unroll
    for (int d = 0; d < 3; ++d) atomicAdd(&ag[d],     w1 * p10[d]);
    #pragma unroll
    for (int d = 0; d < 3; ++d) atomicAdd(&ag[3 + d], w1 * p01[d]);
    #pragma unroll
    for (int d = 0; d < 9; ++d) atomicAdd(&ag[6 + d], w0 * p11[d]);
    #pragma unroll
    for (int d = 0; d < 6; ++d) atomicAdd(&ag[15 + d], w2 * pos[n * 6 + d]);
}

// ---------------------------------------------------------------------------
// K6: per-graph epilogue: Q permutations, 3x3 SVD (Jacobi on A^T A),
// nearest proper rotation (det-fix = reference's Vh row-2 flip), translation.
// Output: [rot G*9 | t G*3 | r_vector(unnormalized) G*9]
// ---------------------------------------------------------------------------
__global__ void k6_epilogue(const float* ws, float* __restrict__ out) {
    int g = blockIdx.x * blockDim.x + threadIdx.x;
    if (g >= GG) return;
    const float* a = ws + WS_ACC + g * 21;
    float out10[3] = {a[0], a[1], a[2]};
    float out01[3] = {a[3], a[4], a[5]};
    float out11[9];
    #pragma unroll
    for (int i = 0; i < 9; ++i) out11[i] = a[6 + i];
    float m1[3] = {a[15], a[16], a[17]};
    float m2[3] = {a[18], a[19], a[20]};

    // Q = [[0,0,1],[1,0,0],[0,1,0]]:  (Qv)[i] = v[(i+2)%3]
    float rvec[9];
    #pragma unroll
    for (int i = 0; i < 3; ++i)
        #pragma unroll
        for (int j = 0; j < 3; ++j)
            rvec[3 * i + j] = out11[3 * ((i + 2) % 3) + ((j + 2) % 3)];
    float tb[3] = {out01[2], out01[0], out01[1]};
    float ta[3] = {out10[2], out10[0], out10[1]};

    // normalize (clipped) and A = transpose(reshape(rnorm,3,3))
    float nrm = 0.f;
    #pragma unroll
    for (int i = 0; i < 9; ++i) nrm += rvec[i] * rvec[i];
    nrm = fmaxf(sqrtf(nrm), 1e-5f);
    float A[3][3];
    #pragma unroll
    for (int i = 0; i < 3; ++i)
        #pragma unroll
        for (int j = 0; j < 3; ++j) A[i][j] = rvec[3 * j + i] / nrm;

    // Jacobi eigen-decomposition of S = A^T A
    float S[3][3], V[3][3] = {{1, 0, 0}, {0, 1, 0}, {0, 0, 1}};
    #pragma unroll
    for (int i = 0; i < 3; ++i)
        #pragma unroll
        for (int j = 0; j < 3; ++j) {
            float s = 0.f;
            #pragma unroll
            for (int k = 0; k < 3; ++k) s += A[k][i] * A[k][j];
            S[i][j] = s;
        }
    for (int sw = 0; sw < 6; ++sw) {
        #pragma unroll
        for (int pi = 0; pi < 3; ++pi) {
            int p = (pi == 2) ? 1 : 0;
            int q = (pi == 0) ? 1 : 2;
            float apq = S[p][q];
            if (fabsf(apq) > 1e-20f) {
                float tau = (S[q][q] - S[p][p]) / (2.f * apq);
                float tt  = ((tau >= 0.f) ? 1.f : -1.f) /
                            (fabsf(tau) + sqrtf(1.f + tau * tau));
                float cc = rsqrtf(1.f + tt * tt);
                float ssn = tt * cc;
                #pragma unroll
                for (int k = 0; k < 3; ++k) {
                    float sp = S[k][p], sq = S[k][q];
                    S[k][p] = cc * sp - ssn * sq;
                    S[k][q] = ssn * sp + cc * sq;
                }
                #pragma unroll
                for (int k = 0; k < 3; ++k) {
                    float sp = S[p][k], sq = S[q][k];
                    S[p][k] = cc * sp - ssn * sq;
                    S[q][k] = ssn * sp + cc * sq;
                }
                #pragma unroll
                for (int k = 0; k < 3; ++k) {
                    float vp = V[k][p], vq = V[k][q];
                    V[k][p] = cc * vp - ssn * vq;
                    V[k][q] = ssn * vp + cc * vq;
                }
            }
        }
    }
    // sort eigenvalues descending (swap V columns)
    float ev[3] = {S[0][0], S[1][1], S[2][2]};
    #pragma unroll
    for (int i = 0; i < 2; ++i)
        #pragma unroll
        for (int j = 0; j < 2 - i; ++j)
            if (ev[j] < ev[j + 1]) {
                float t0 = ev[j]; ev[j] = ev[j + 1]; ev[j + 1] = t0;
                #pragma unroll
                for (int k = 0; k < 3; ++k) {
                    float tv = V[k][j]; V[k][j] = V[k][j + 1]; V[k][j + 1] = tv;
                }
            }
    // U = A V, columns normalized
    float U[3][3];
    #pragma unroll
    for (int j = 0; j < 3; ++j) {
        float s2 = 0.f;
        #pragma unroll
        for (int i = 0; i < 3; ++i) {
            float u = 0.f;
            #pragma unroll
            for (int k = 0; k < 3; ++k) u += A[i][k] * V[k][j];
            U[i][j] = u;
            s2 += u * u;
        }
        float inv = rsqrtf(fmaxf(s2, 1e-24f));
        #pragma unroll
        for (int i = 0; i < 3; ++i) U[i][j] *= inv;
    }
    // R = U V^T; force det(R) = +1 (flip direction of smallest sigma)
    float R[3][3];
    #pragma unroll
    for (int i = 0; i < 3; ++i)
        #pragma unroll
        for (int j = 0; j < 3; ++j) {
            float s = 0.f;
            #pragma unroll
            for (int k = 0; k < 3; ++k) s += U[i][k] * V[j][k];
            R[i][j] = s;
        }
    float det = R[0][0] * (R[1][1] * R[2][2] - R[1][2] * R[2][1])
              - R[0][1] * (R[1][0] * R[2][2] - R[1][2] * R[2][0])
              + R[0][2] * (R[1][0] * R[2][1] - R[1][1] * R[2][0]);
    if (det < 0.f) {
        #pragma unroll
        for (int i = 0; i < 3; ++i)
            #pragma unroll
            for (int j = 0; j < 3; ++j) R[i][j] -= 2.f * U[i][2] * V[j][2];
    }
    // t = m2 + trans_b - R(m1 + trans_a)
    float tvec[3];
    #pragma unroll
    for (int i = 0; i < 3; ++i) {
        float s = 0.f;
        #pragma unroll
        for (int k = 0; k < 3; ++k) s += R[i][k] * (m1[k] + ta[k]);
        tvec[i] = m2[i] + tb[i] - s;
    }
    #pragma unroll
    for (int i = 0; i < 3; ++i)
        #pragma unroll
        for (int j = 0; j < 3; ++j) out[g * 9 + i * 3 + j] = R[i][j];
    #pragma unroll
    for (int i = 0; i < 3; ++i) out[GG * 9 + g * 3 + i] = tvec[i];
    #pragma unroll
    for (int i = 0; i < 9; ++i) out[GG * 12 + g * 9 + i] = rvec[i];
}

// ---------------------------------------------------------------------------
// Launcher
// ---------------------------------------------------------------------------
extern "C" void kernel_launch(void* const* d_in, const int* in_sizes, int n_in,
                              void* d_out, int out_size, void* d_ws, size_t ws_size,
                              hipStream_t stream) {
    const float* x00   = (const float*)d_in[0];
    const float* x10   = (const float*)d_in[1];
    const float* x01   = (const float*)d_in[2];
    const float* x11   = (const float*)d_in[3];
    const float* pos   = (const float*)d_in[4];
    const int*   gid   = (const int*)d_in[5];
    const float* gamma = (const float*)d_in[7];
    const float* beta  = (const float*)d_in[8];
    const float* lin_w = (const float*)d_in[9];
    const float* lin_b = (const float*)d_in[10];
    const float* w10   = (const float*)d_in[11];
    const float* w01   = (const float*)d_in[12];
    const float* w11   = (const float*)d_in[13];
    float* ws  = (float*)d_ws;
    float* out = (float*)d_out;

    k0_init<<<(WS_TOTAL + 255) / 256, 256, 0, stream>>>(ws);
    k1_stats<<<(NN * 16) / 256, 256, 0, stream>>>(x00, gid, ws);
    k2_finalize<<<(GG * 64) / 256, 256, 0, stream>>>(gamma, beta, ws);
    // one wave (32 lanes) per 16-node tile, 8 waves per block
    k3_logits_wmma<<<(NN / 16) / 8, 256, 0, stream>>>(x00, gid, lin_w, lin_b, ws);
    k4_exp<<<(NN * 3) / 256, 256, 0, stream>>>(gid, ws);
    k5_accumulate<<<NN / 256, 256, 0, stream>>>(x10, x01, x11, pos, gid,
                                                w10, w01, w11, ws);
    k6_epilogue<<<(GG + 255) / 256, 256, 0, stream>>>(ws, out);
}
